// LigerLMHeadCE_69217692942470
// MI455X (gfx1250) — compile-verified, matrix operand order checked by reading, not run on hardware
//
#include <hip/hip_runtime.h>
#include <math.h>

// Problem constants (match reference)
#define N_TOK   2048
#define HDIM    2048
#define VOCAB   128000
#define IGNORE_INDEX (-100)

// Tiling
#define NT      128                 // tokens per block tile
#define VT      128                 // vocab rows per GEMM tile
#define KT      32                  // bf16 WMMA K
#define NSLICE  50                  // vocab slices (grid.y)
#define VPS     (VOCAB / NSLICE)    // 2560 vocab entries per slice
#define VTILES  (VPS / VT)          // 20 tiles per slice
#define NGRP    (N_TOK / NT)        // 16 token groups (grid.x)
#define LPITCH  40                  // ushorts per LDS tile row (32 data + 8 pad -> 80B)

typedef __attribute__((ext_vector_type(16))) __bf16 v16bf;
typedef __attribute__((ext_vector_type(8)))  float  v8f;
typedef unsigned short u16;
typedef unsigned int   u32;
typedef unsigned long long u64;

union Frag { v16bf v; uint4 q[2]; };

// ---- CDNA5 async global->LDS copy (no VGPR staging; tracked by ASYNCcnt) ----
__device__ __forceinline__ void async_b128(u32 lds_addr, const void* gaddr) {
  asm volatile("global_load_async_to_lds_b128 %0, %1, off"
               :: "v"(lds_addr), "v"((u64)(size_t)gaddr)
               : "memory");
}
__device__ __forceinline__ void wait_async0() {
  asm volatile("s_wait_asynccnt 0x0" ::: "memory");
}

// RNE pack (used once, in the bandwidth-bound convert pass)
__device__ __forceinline__ u32 pack_bf16_rne(float a, float b) {
  u32 ua = __float_as_uint(a);
  u32 ub = __float_as_uint(b);
  ua = (ua + 0x7FFFu + ((ua >> 16) & 1u)) >> 16;
  ub = (ub + 0x7FFFu + ((ub >> 16) & 1u)) >> 16;
  return ua | (ub << 16);
}

// Truncating pack: single v_perm_b32 (fallback path inner loop)
__device__ __forceinline__ u32 pack_bf16_trunc(float a, float b) {
  return __builtin_amdgcn_perm(__float_as_uint(b), __float_as_uint(a), 0x07060302u);
}

// Safe (max, sumexp) merge; handles -inf identities without NaN.
__device__ __forceinline__ void merge_ms(float& m, float& s, float m2, float s2) {
  float mn = fmaxf(m, m2);
  float e1 = (m  == mn) ? 1.0f : __expf(m  - mn);
  float e2 = (m2 == mn) ? 1.0f : __expf(m2 - mn);
  s = s * e1 + s2 * e2;
  m = mn;
}

// ---------------------------------------------------------------------------
// Kernel 0: f32 -> bf16 (RNE) bulk convert, 8 elements/thread
// ---------------------------------------------------------------------------
__global__ __launch_bounds__(256)
void cvt_f32_bf16(const float* __restrict__ src, u16* __restrict__ dst) {
  const size_t i = ((size_t)blockIdx.x * 256 + threadIdx.x) * 8;
  float4 a = *(const float4*)(src + i);
  float4 b = *(const float4*)(src + i + 4);
  uint4 p;
  p.x = pack_bf16_rne(a.x, a.y); p.y = pack_bf16_rne(a.z, a.w);
  p.z = pack_bf16_rne(b.x, b.y); p.w = pack_bf16_rne(b.z, b.w);
  *(uint4*)(dst + i) = p;
}

// ---------------------------------------------------------------------------
// Kernel A: bf16 WMMA GEMM (W slice @ x^T) fused with online softmax partials.
// grid = (NGRP, NSLICE), block = 256 threads (8 waves).
// PRE=true : double-buffered LDS fed by global_load_async_to_lds_b128.
// PRE=false: single-buffer, two-barrier staging with v_perm bf16 pack.
// ---------------------------------------------------------------------------
template<bool PRE>
__global__ __launch_bounds__(256)
void lmhead_ce_partial(const float* __restrict__ xf, const float* __restrict__ Wf,
                       const u16* __restrict__ x16, const u16* __restrict__ W16,
                       const float* __restrict__ bias, const int* __restrict__ y,
                       float* __restrict__ pm, float* __restrict__ ps,
                       float* __restrict__ tgt) {
  __shared__ __align__(16) u16 sW[2][VT * LPITCH];
  __shared__ __align__(16) u16 sX[2][NT * LPITCH];
  __shared__ __align__(16) float sBias[VT];
  __shared__ int   sY[NT];
  __shared__ float redM[2][4][64];
  __shared__ float redS[2][4][64];
  __shared__ float rowM[NT];
  __shared__ float rowS[NT];

  const int tid  = threadIdx.x;
  const int lane = tid & 31;
  const int wave = tid >> 5;
  const int wv   = wave & 3;            // vocab quadrant
  const int wn   = wave >> 2;           // token half
  const int vv   = wv * 32;
  const int nn   = wn * 64;
  const int lr   = lane & 15;
  const int khalf = (lane >> 4) << 3;   // ISA 16-bit A/B fragment K layout
  const int hi8   = (lane >> 4) << 3;   // C-layout upper-half M offset

  const int n0  = blockIdx.x * NT;
  const int vs0 = blockIdx.y * VPS;

  if (tid < NT) { rowM[tid] = -INFINITY; rowS[tid] = 0.0f; sY[tid] = y[n0 + tid]; }

  const int ldrow  = tid >> 1;          // 0..127
  const int ldhalf = (tid & 1) << 4;    // element offset 0 or 16
  const int ldsoff = ldrow * LPITCH + ldhalf;

  const float* gXf = xf + (size_t)(n0 + ldrow) * HDIM + ldhalf;
  const u16*   gXh = x16 + (size_t)(n0 + ldrow) * HDIM + ldhalf;

  // Per-thread LDS byte addresses for async staging (aperture low bits = offset)
  const u32 aW0 = (u32)(size_t)&sW[0][ldsoff];
  const u32 aW1 = (u32)(size_t)&sW[1][ldsoff];
  const u32 aX0 = (u32)(size_t)&sX[0][ldsoff];
  const u32 aX1 = (u32)(size_t)&sX[1][ldsoff];

  for (int vt = 0; vt < VTILES; ++vt) {
    const int vbase = vs0 + vt * VT;
    if (tid < VT) sBias[tid] = bias[vbase + tid];
    const float* gWf = Wf + (size_t)(vbase + ldrow) * HDIM + ldhalf;
    const u16*   gWh = W16 + (size_t)(vbase + ldrow) * HDIM + ldhalf;

    v8f acc[2][4];
    {
      v8f z = {};
      #pragma unroll
      for (int i = 0; i < 2; ++i)
        #pragma unroll
        for (int j = 0; j < 4; ++j) acc[i][j] = z;
    }

    // One K-step of fragment loads + 8 WMMAs from buffer b
    auto mma_step = [&](int b) {
      Frag fa[2], fb[4];
      #pragma unroll
      for (int tv = 0; tv < 2; ++tv) {
        const u16* p = &sW[b][(vv + tv * 16 + lr) * LPITCH + khalf];
        fa[tv].q[0] = *(const uint4*)p;
        fa[tv].q[1] = *(const uint4*)(p + 16);
      }
      #pragma unroll
      for (int tn = 0; tn < 4; ++tn) {
        const u16* p = &sX[b][(nn + tn * 16 + lr) * LPITCH + khalf];
        fb[tn].q[0] = *(const uint4*)p;
        fb[tn].q[1] = *(const uint4*)(p + 16);
      }
      #pragma unroll
      for (int tv = 0; tv < 2; ++tv)
        #pragma unroll
        for (int tn = 0; tn < 4; ++tn)
          acc[tv][tn] = __builtin_amdgcn_wmma_f32_16x16x32_bf16(
              false, fa[tv].v, false, fb[tn].v, (short)0, acc[tv][tn], false, false);
    };

    if (PRE) {
      // ---- async-DMA pipeline: zero staging registers ----
      async_b128(aW0,      gWh);     async_b128(aW0 + 16, gWh + 8);
      async_b128(aX0,      gXh);     async_b128(aX0 + 16, gXh + 8);
      int buf = 0;
      for (int kk = 0; kk < HDIM; kk += KT) {
        wait_async0();               // this wave's async fills of buf done
        __syncthreads();             // all waves filled buf; buf^1 readers done
        if (kk + KT < HDIM) {        // issue next K-step DMA into buf^1
          const u32 dW = buf ? aW0 : aW1;
          const u32 dX = buf ? aX0 : aX1;
          const u16* w  = gWh + kk + KT;
          const u16* xx = gXh + kk + KT;
          async_b128(dW, w);  async_b128(dW + 16, w + 8);
          async_b128(dX, xx); async_b128(dX + 16, xx + 8);
        }
        if (kk + 2 * KT < HDIM)
          __builtin_prefetch(gWh + kk + 2 * KT, 0, 3);
        mma_step(buf);
        buf ^= 1;
      }
    } else {
      // ---- fallback: single buffer, inline v_perm conversion ----
      for (int kk = 0; kk < HDIM; kk += KT) {
        __syncthreads();             // previous frag reads done
        {
          const float4* g = (const float4*)(gWf + kk);
          float4 a0 = g[0], a1 = g[1], a2 = g[2], a3 = g[3];
          uint4 p0, p1;
          p0.x = pack_bf16_trunc(a0.x, a0.y); p0.y = pack_bf16_trunc(a0.z, a0.w);
          p0.z = pack_bf16_trunc(a1.x, a1.y); p0.w = pack_bf16_trunc(a1.z, a1.w);
          p1.x = pack_bf16_trunc(a2.x, a2.y); p1.y = pack_bf16_trunc(a2.z, a2.w);
          p1.z = pack_bf16_trunc(a3.x, a3.y); p1.w = pack_bf16_trunc(a3.z, a3.w);
          *(uint4*)&sW[0][ldsoff] = p0; *(uint4*)&sW[0][ldsoff + 8] = p1;
        }
        {
          const float4* g = (const float4*)(gXf + kk);
          float4 a0 = g[0], a1 = g[1], a2 = g[2], a3 = g[3];
          uint4 p0, p1;
          p0.x = pack_bf16_trunc(a0.x, a0.y); p0.y = pack_bf16_trunc(a0.z, a0.w);
          p0.z = pack_bf16_trunc(a1.x, a1.y); p0.w = pack_bf16_trunc(a1.z, a1.w);
          p1.x = pack_bf16_trunc(a2.x, a2.y); p1.y = pack_bf16_trunc(a2.z, a2.w);
          p1.z = pack_bf16_trunc(a3.x, a3.y); p1.w = pack_bf16_trunc(a3.z, a3.w);
          *(uint4*)&sX[0][ldsoff] = p0; *(uint4*)&sX[0][ldsoff + 8] = p1;
        }
        __syncthreads();
        mma_step(0);
      }
    }

    // ---- epilogue: bias, target peel, per-column (max, sumexp) over 128 vocab
    float bv[16];
    #pragma unroll
    for (int tv = 0; tv < 2; ++tv) {
      float4 b0 = *(const float4*)&sBias[vv + tv * 16 + hi8];
      float4 b1 = *(const float4*)&sBias[vv + tv * 16 + hi8 + 4];
      bv[tv * 8 + 0] = b0.x; bv[tv * 8 + 1] = b0.y;
      bv[tv * 8 + 2] = b0.z; bv[tv * 8 + 3] = b0.w;
      bv[tv * 8 + 4] = b1.x; bv[tv * 8 + 5] = b1.y;
      bv[tv * 8 + 6] = b1.z; bv[tv * 8 + 7] = b1.w;
    }
    #pragma unroll
    for (int tn = 0; tn < 4; ++tn) {
      const int col = nn + tn * 16 + lr;
      const int tok = n0 + col;
      const int yv  = sY[col];
      float m = -INFINITY;
      float tval = 0.0f;
      bool  found = false;
      #pragma unroll
      for (int tv = 0; tv < 2; ++tv)
        #pragma unroll
        for (int r = 0; r < 8; ++r) {
          const int vloc = vv + tv * 16 + hi8 + r;      // C layout (ISA 7.12.2)
          float val = acc[tv][tn][r] + bv[tv * 8 + r];
          const bool hit = (vbase + vloc) == yv;
          tval  = hit ? val : tval;
          found = found || hit;
          m = fmaxf(m, val);
        }
      if (found) tgt[tok] = tval;            // single predicated store per group
      float s = 0.0f;
      #pragma unroll
      for (int tv = 0; tv < 2; ++tv)
        #pragma unroll
        for (int r = 0; r < 8; ++r)
          s += __expf((acc[tv][tn][r] + bv[tv * 8 + r]) - m);
      float m2 = __shfl_xor(m, 16);
      float s2 = __shfl_xor(s, 16);
      merge_ms(m, s, m2, s2);
      if (lane < 16) { redM[wn][wv][tn * 16 + lr] = m; redS[wn][wv][tn * 16 + lr] = s; }
    }
    __syncthreads();
    if (tid < NT) {
      const int gn = tid >> 6, c = tid & 63;
      float m = rowM[tid], s = rowS[tid];
      #pragma unroll
      for (int q = 0; q < 4; ++q) merge_ms(m, s, redM[gn][q][c], redS[gn][q][c]);
      rowM[tid] = m; rowS[tid] = s;
    }
    __syncthreads();
  }

  if (tid < NT) {
    pm[(size_t)blockIdx.y * N_TOK + n0 + tid] = rowM[tid];
    ps[(size_t)blockIdx.y * N_TOK + n0 + tid] = rowS[tid];
  }
}

// ---------------------------------------------------------------------------
// Kernel B: per-row merge of NSLICE partials -> nll[row]
// ---------------------------------------------------------------------------
__global__ __launch_bounds__(64)
void lmhead_ce_lse(const float* __restrict__ pm, const float* __restrict__ ps,
                   const float* __restrict__ tgt, const int* __restrict__ y,
                   float* __restrict__ nll) {
  const int r = blockIdx.x;
  const int t = threadIdx.x;
  float m = -INFINITY, s = 0.0f;
  if (t < NSLICE) { m = pm[(size_t)t * N_TOK + r]; s = ps[(size_t)t * N_TOK + r]; }
  __shared__ float sm[64], ss[64];
  sm[t] = m; ss[t] = s;
  __syncthreads();
  for (int off = 32; off > 0; off >>= 1) {
    if (t < off) {
      float mm = sm[t], sv = ss[t];
      merge_ms(mm, sv, sm[t + off], ss[t + off]);
      sm[t] = mm; ss[t] = sv;
    }
    __syncthreads();
  }
  if (t == 0) {
    const int yv = y[r];
    nll[r] = (yv != IGNORE_INDEX) ? (sm[0] + logf(ss[0]) - tgt[r]) : 0.0f;
  }
}

// ---------------------------------------------------------------------------
// Kernel C: mean over valid rows -> scalar loss
// ---------------------------------------------------------------------------
__global__ __launch_bounds__(256)
void lmhead_ce_reduce(const float* __restrict__ nll, const int* __restrict__ y,
                      float* __restrict__ out) {
  const int t = threadIdx.x;
  float sum = 0.0f, cnt = 0.0f;
  for (int r = t; r < N_TOK; r += 256)
    if (y[r] != IGNORE_INDEX) { sum += nll[r]; cnt += 1.0f; }
  __shared__ float s1[256], s2[256];
  s1[t] = sum; s2[t] = cnt;
  __syncthreads();
  for (int off = 128; off > 0; off >>= 1) {
    if (t < off) { s1[t] += s1[t + off]; s2[t] += s2[t + off]; }
    __syncthreads();
  }
  if (t == 0) out[0] = s1[0] / fmaxf(s2[0], 1.0f);
}

extern "C" void kernel_launch(void* const* d_in, const int* in_sizes, int n_in,
                              void* d_out, int out_size, void* d_ws, size_t ws_size,
                              hipStream_t stream) {
  (void)in_sizes; (void)n_in; (void)out_size;
  const float* x  = (const float*)d_in[0];
  const float* W  = (const float*)d_in[1];
  const float* b  = (const float*)d_in[2];
  const int*   y  = (const int*)d_in[3];

  // Workspace layout: small reductions first (fallback-safe), bf16 mirrors after.
  float* wsf = (float*)d_ws;
  float* pm  = wsf;                               // NSLICE * N_TOK
  float* ps  = pm + (size_t)NSLICE * N_TOK;       // NSLICE * N_TOK
  float* tgt = ps + (size_t)NSLICE * N_TOK;       // N_TOK
  float* nll = tgt + N_TOK;                       // N_TOK
  const size_t small_bytes = ((size_t)NSLICE * N_TOK * 2 + 2 * N_TOK) * sizeof(float);
  u16* x16 = (u16*)((char*)d_ws + small_bytes);   // N_TOK * HDIM bf16
  u16* W16 = x16 + (size_t)N_TOK * HDIM;          // VOCAB * HDIM bf16
  const size_t need_fast = small_bytes +
      ((size_t)N_TOK * HDIM + (size_t)VOCAB * HDIM) * sizeof(u16);

  dim3 gridA(NGRP, NSLICE);
  if (ws_size >= need_fast) {
    cvt_f32_bf16<<<(unsigned)(((size_t)N_TOK * HDIM) / (256 * 8)), 256, 0, stream>>>(x, x16);
    cvt_f32_bf16<<<(unsigned)(((size_t)VOCAB * HDIM) / (256 * 8)), 256, 0, stream>>>(W, W16);
    lmhead_ce_partial<true><<<gridA, 256, 0, stream>>>(x, W, x16, W16, b, y, pm, ps, tgt);
  } else {
    lmhead_ce_partial<false><<<gridA, 256, 0, stream>>>(x, W, x16, W16, b, y, pm, ps, tgt);
  }
  lmhead_ce_lse<<<N_TOK, 64, 0, stream>>>(pm, ps, tgt, y, nll);
  lmhead_ce_reduce<<<1, 256, 0, stream>>>(nll, y, (float*)d_out);
}
